// NCPModel_81346680586935
// MI455X (gfx1250) — matrix operand order repaired and checked by reading, not compiled, not run on hardware
//
#include <hip/hip_runtime.h>
#include <hip/hip_bf16.h>
#include <stdint.h>

// ---------------------------------------------------------------------------
// NCP / CfC recurrent network, MI455X (gfx1250, wave32, WMMA bf16 path)
//   - 32 workgroups, each owns a 16-row batch tile for the whole T loop
//   - recurrent state lives in LDS as bf16 (A-operand layout-friendly)
//   - weights packed once to padded bf16 [h_pad, cat_pad]; streamed from L2
//   - v_wmma_f32_16x16x32_bf16, 4 accumulators/wave (ff1, ff2, ta, tb)
//   - epilogue uses hw transcendentals (v_exp_f32 / v_rcp_f32) instead of
//     library tanhf to keep the scalar tail off the critical path
// ---------------------------------------------------------------------------

typedef __attribute__((ext_vector_type(16))) __bf16 v16bf;
typedef __attribute__((ext_vector_type(8)))  float  v8f;

struct Q2 { uint4 a, b; };   // 32 bytes == one v16bf fragment

#define BATCH  512
#define TSTEPS 256

// layer dims (fi: 64,116,76 ; h: 116,76,64 ; cat = fi+h)
#define H0 116
#define H1 76
#define H2 64
#define CATP0 192   // cat 180 -> pad 192 (6 k-tiles of 32)
#define CATP1 192   // cat 192 exact
#define CATP2 160   // cat 140 -> pad 160 (5 k-tiles)
#define HPAD0 128
#define HPAD1 80
#define HPAD2 64
#define NKT0 6
#define NKT1 6
#define NKT2 5

// workspace layout (ushort elements)
#define OFF_L0 0
#define OFF_L1 (OFF_L0 + 4 * HPAD0 * CATP0)         // 98304
#define OFF_L2 (OFF_L1 + 4 * HPAD1 * CATP1)         // 159744
#define OFF_FC (OFF_L2 + 4 * HPAD2 * CATP2)         // 200704

__device__ __forceinline__ unsigned short f2bf(float f) {
  union { float f; unsigned int u; } c; c.f = f;
  unsigned int u = c.u;
  u += 0x7FFFu + ((u >> 16) & 1u);                  // round-to-nearest-even
  return (unsigned short)(u >> 16);
}

// tanh via hw exp/rcp: tanh(x) = sign(x) * (1 - 2/(exp(2|x|)+1)).
// exp arg >= 0, so overflow -> +inf -> rcp -> 0 -> result saturates to +-1.
__device__ __forceinline__ float fast_tanh(float x) {
  float e = __expf(2.0f * fabsf(x));                // v_exp_f32
  float r = 1.0f - 2.0f * __builtin_amdgcn_rcpf(e + 1.0f);  // v_rcp_f32
  return copysignf(r, x);
}

__device__ __forceinline__ float fast_sigmoid(float x) {
  return __builtin_amdgcn_rcpf(1.0f + __expf(-x));
}

// A fragment (16x32 bf16, M=batch rows): lane<16 holds row=lane, K {0..7,16..23};
// lane>=16 holds row=lane-16, K {8..15,24..31}. Two 16B LDS loads per lane.
__device__ __forceinline__ v16bf load_a_frag(const unsigned short* cat, int catp,
                                             int lane, int k0) {
  const int row  = lane & 15;
  const int koff = (lane >> 4) << 3;
  const unsigned short* p = cat + row * catp + k0 + koff;
  Q2 q;
  q.a = *(const uint4*)(p);        // K = k0+koff   .. +7
  q.b = *(const uint4*)(p + 16);   // K = k0+koff+16 .. +23
  return __builtin_bit_cast(v16bf, q);
}

// B fragment (32x16 bf16, N=output features): lane<16 = column n0+lane, K {0..7,16..23};
// lane>=16 = column n0+lane-16, K {8..15,24..31}. W stored [h_pad, cat_pad] row-major,
// so each lane reads two 16-byte chunks of its W row, starting 32 bytes apart.
__device__ __forceinline__ v16bf load_b_frag(const unsigned short* __restrict__ wb,
                                             int catp, int lane, int n0, int k0) {
  const int nrow = n0 + (lane & 15);
  const int kc   = k0 + ((lane >> 4) << 3);
  const unsigned short* p = wb + (size_t)nrow * catp + kc;
  Q2 q;
  q.a = *(const uint4*)(p);        // K = kc .. kc+7
  q.b = *(const uint4*)(p + 16);   // K = kc+16 .. kc+23
  return __builtin_bit_cast(v16bf, q);
}

// One CfC cell for this workgroup's 16-row tile. Wave w handles N-tile w.
// Barriers are executed unconditionally by all waves.
__device__ __forceinline__ void cfc_layer(
    const unsigned short* __restrict__ catb, int catp,
    const unsigned short* __restrict__ wb, int hpad, int nkt,
    int h, int ntiles, int w, int lane,
    float bb1, float bb2, float bt,
    unsigned short* __restrict__ o1, int o1p, int o1c,   // own recurrent slot
    unsigned short* __restrict__ o2, int o2p, int o2c)   // next layer's input slot
{
  float hv[8];
  bool valid = false;
  int n = 0, mbase = 0;
  if (w < ntiles) {
    v8f a0 = {0,0,0,0,0,0,0,0}, a1 = a0, a2 = a0, a3 = a0;
    const int n0  = w << 4;
    const int msz = hpad * catp;
    for (int kt = 0; kt < nkt; ++kt) {
      const int k0 = kt << 5;
      v16bf av  = load_a_frag(catb, catp, lane, k0);
      v16bf b0v = load_b_frag(wb,            catp, lane, n0, k0);
      a0 = __builtin_amdgcn_wmma_f32_16x16x32_bf16(false, av, false, b0v, (short)0, a0, false, false);
      v16bf b1v = load_b_frag(wb +     msz,  catp, lane, n0, k0);
      a1 = __builtin_amdgcn_wmma_f32_16x16x32_bf16(false, av, false, b1v, (short)0, a1, false, false);
      v16bf b2v = load_b_frag(wb + 2 * msz,  catp, lane, n0, k0);
      a2 = __builtin_amdgcn_wmma_f32_16x16x32_bf16(false, av, false, b2v, (short)0, a2, false, false);
      v16bf b3v = load_b_frag(wb + 3 * msz,  catp, lane, n0, k0);
      a3 = __builtin_amdgcn_wmma_f32_16x16x32_bf16(false, av, false, b3v, (short)0, a3, false, false);
    }
    n     = n0 + (lane & 15);
    mbase = (lane >> 4) << 3;                  // D layout: lanes 0-15 rows 0-7, 16-31 rows 8-15
    valid = n < h;
    for (int r = 0; r < 8; ++r) {
      float ti = fast_sigmoid(a2[r] + a3[r] + bt);
      float f1 = fast_tanh(a0[r] + bb1);
      float f2 = fast_tanh(a1[r] + bb2);
      hv[r] = f1 * (1.0f - ti) + ti * f2;
    }
  }
  __syncthreads();                             // all A reads done before state update
  if (valid) {
    for (int r = 0; r < 8; ++r) {
      unsigned short hb = f2bf(hv[r]);
      int m = mbase + r;
      o1[m * o1p + o1c + n] = hb;
      o2[m * o2p + o2c + n] = hb;
    }
  }
  __syncthreads();                             // state visible to next layer
}

__global__ __launch_bounds__(256, 1) void ncp_forward(
    const float* __restrict__ x, float* __restrict__ out,
    const unsigned short* __restrict__ ws,
    const float* __restrict__ b1_0, const float* __restrict__ b2_0,
    const float* __restrict__ ba_0, const float* __restrict__ bb_0,
    const float* __restrict__ b1_1, const float* __restrict__ b2_1,
    const float* __restrict__ ba_1, const float* __restrict__ bb_1,
    const float* __restrict__ b1_2, const float* __restrict__ b2_2,
    const float* __restrict__ ba_2, const float* __restrict__ bb_2,
    const float* __restrict__ bfc)
{
  __shared__ __align__(16) unsigned short cat0[16 * CATP0];  // [x_t | h0]
  __shared__ __align__(16) unsigned short cat1[16 * CATP1];  // [h0  | h1]
  __shared__ __align__(16) unsigned short cat2[16 * CATP2];  // [h1  | h2]
  __shared__ __align__(16) unsigned short h2b [16 * 64];     // aligned copy of h2 for FC

  const int tid  = threadIdx.x;
  const int w    = tid >> 5;
  const int lane = tid & 31;
  const int b0   = blockIdx.x << 4;

  for (int i = tid; i < 16 * CATP0; i += 256) { cat0[i] = 0; cat1[i] = 0; }
  for (int i = tid; i < 16 * CATP2; i += 256) cat2[i] = 0;
  for (int i = tid; i < 16 * 64;    i += 256) h2b[i]  = 0;
  __syncthreads();

  // hoist per-wave biases out of the time loop
  const int nA = (w << 4) + (lane & 15);
  float l0b1 = 0, l0b2 = 0, l0bt = 0;
  float l1b1 = 0, l1b2 = 0, l1bt = 0;
  float l2b1 = 0, l2b2 = 0, l2bt = 0, fcb = 0;
  if (nA < H0)          { l0b1 = b1_0[nA]; l0b2 = b2_0[nA]; l0bt = ba_0[nA] + bb_0[nA]; }
  if (w < 5 && nA < H1) { l1b1 = b1_1[nA]; l1b2 = b2_1[nA]; l1bt = ba_1[nA] + bb_1[nA]; }
  if (w < 4)            { l2b1 = b1_2[nA]; l2b2 = b2_2[nA]; l2bt = ba_2[nA] + bb_2[nA];
                          fcb  = bfc[nA]; }

  for (int t = 0; t < TSTEPS; ++t) {
    { // stage x_t [16,64] -> bf16 into cat0 cols 0..63 (256 threads x float4)
      const int m = tid >> 4;
      const int c = (tid & 15) << 2;
      float4 xv = *(const float4*)(x + (((size_t)(b0 + m)) * TSTEPS + t) * 64 + c);
      unsigned short* dst = cat0 + m * CATP0 + c;
      dst[0] = f2bf(xv.x); dst[1] = f2bf(xv.y); dst[2] = f2bf(xv.z); dst[3] = f2bf(xv.w);
    }
    __syncthreads();

    // inter -> command -> motor
    cfc_layer(cat0, CATP0, ws + OFF_L0, HPAD0, NKT0, H0, 8, w, lane,
              l0b1, l0b2, l0bt, cat0, CATP0, 64, cat1, CATP1, 0);
    cfc_layer(cat1, CATP1, ws + OFF_L1, HPAD1, NKT1, H1, 5, w, lane,
              l1b1, l1b2, l1bt, cat1, CATP1, H0, cat2, CATP2, 0);
    cfc_layer(cat2, CATP2, ws + OFF_L2, HPAD2, NKT2, H2, 4, w, lane,
              l2b1, l2b2, l2bt, cat2, CATP2, H1, h2b, 64, 0);

    // fused output FC: y[16,64] = h2 @ w_fc^T + b_fc
    if (w < 4) {
      v8f acc = {0,0,0,0,0,0,0,0};
      const unsigned short* wfc = ws + OFF_FC;
      const int n0 = w << 4;
      for (int kt = 0; kt < 2; ++kt) {
        v16bf av = load_a_frag(h2b, 64, lane, kt << 5);
        v16bf bv = load_b_frag(wfc, 64, lane, n0, kt << 5);
        acc = __builtin_amdgcn_wmma_f32_16x16x32_bf16(false, av, false, bv, (short)0, acc, false, false);
      }
      const int mb = (lane >> 4) << 3;
      float* op = out + (((size_t)(b0 + mb)) * TSTEPS + t) * 64 + nA;
      for (int r = 0; r < 8; ++r)
        op[(size_t)r * TSTEPS * 64] = acc[r] + fcb;
    }
  }
}

// Pack f32 weights (optionally masked) into padded bf16 [h_pad, cat_pad] row-major.
__global__ void pack_weights(const float* __restrict__ wsrc, const float* __restrict__ mask,
                             unsigned short* __restrict__ dst,
                             int h, int cat, int catp, int total)
{
  int i = blockIdx.x * blockDim.x + threadIdx.x;
  if (i >= total) return;
  int nrow = i / catp;
  int k    = i - nrow * catp;
  float v = 0.0f;
  if (nrow < h && k < cat) {
    v = wsrc[nrow * cat + k];
    if (mask) v *= mask[nrow * cat + k];
  }
  dst[i] = f2bf(v);
}

extern "C" void kernel_launch(void* const* d_in, const int* in_sizes, int n_in,
                              void* d_out, int out_size, void* d_ws, size_t ws_size,
                              hipStream_t stream)
{
  (void)in_sizes; (void)n_in; (void)out_size; (void)ws_size;

  // setup_inputs() dict order:
  // 0:mask_0, 1..8: (w,b) x (ff1,ff2,ta,tb) layer0, 9:mask_1, 10..17: layer1,
  // 18:mask_2, 19..26: layer2, 27:w_fc, 28:b_fc, 29:x
  const float* mask[3] = { (const float*)d_in[0], (const float*)d_in[9], (const float*)d_in[18] };
  const float* W[3][4];
  const float* Bv[3][4];
  for (int l = 0; l < 3; ++l) {
    const int base = l * 9 + 1;
    for (int m = 0; m < 4; ++m) {
      W[l][m]  = (const float*)d_in[base + 2 * m];
      Bv[l][m] = (const float*)d_in[base + 2 * m + 1];
    }
  }
  const float* wfc = (const float*)d_in[27];
  const float* bfc = (const float*)d_in[28];
  const float* x   = (const float*)d_in[29];
  float* out = (float*)d_out;
  unsigned short* ws = (unsigned short*)d_ws;

  const int hh[3]   = { H0, H1, H2 };
  const int cc[3]   = { 180, 192, 140 };
  const int catp[3] = { CATP0, CATP1, CATP2 };
  const int hpad[3] = { HPAD0, HPAD1, HPAD2 };
  const size_t off[3] = { OFF_L0, OFF_L1, OFF_L2 };

  for (int l = 0; l < 3; ++l) {
    const int per = hpad[l] * catp[l];
    for (int m = 0; m < 4; ++m) {
      const float* mk = (m < 2) ? mask[l] : nullptr;   // mask only ff1/ff2
      pack_weights<<<(per + 255) / 256, 256, 0, stream>>>(
          W[l][m], mk, ws + off[l] + (size_t)m * per, hh[l], cc[l], catp[l], per);
    }
  }
  pack_weights<<<(64 * 64 + 255) / 256, 256, 0, stream>>>(
      wfc, nullptr, ws + OFF_FC, 64, 64, 64, 64 * 64);

  ncp_forward<<<BATCH / 16, 256, 0, stream>>>(
      x, out, ws,
      Bv[0][0], Bv[0][1], Bv[0][2], Bv[0][3],
      Bv[1][0], Bv[1][1], Bv[1][2], Bv[1][3],
      Bv[2][0], Bv[2][1], Bv[2][2], Bv[2][3],
      bfc);
}
